// MultiHeadAttention_65773129171094
// MI455X (gfx1250) — matrix operand order repaired and checked by reading, not compiled, not run on hardware
//
#include <hip/hip_runtime.h>
#include <hip/hip_bf16.h>
#include <stdint.h>

// ---------- types matching the CDNA5 WMMA / TDM builtin signatures ----------
typedef __attribute__((ext_vector_type(16))) __bf16 v16bf;
typedef __attribute__((ext_vector_type(8)))  __bf16 v8bf;
typedef __attribute__((ext_vector_type(4)))  __bf16 v4bf;
typedef __attribute__((ext_vector_type(8)))  float  v8f;
typedef __attribute__((ext_vector_type(4)))  unsigned int u32x4;
typedef __attribute__((ext_vector_type(8)))  int i32x8;
typedef __attribute__((ext_vector_type(4)))  int i32x4;

#define N_HIDDEN 1024
#define N_HEAD   16
#define DIM_HEAD 64
#define BATCH    8
#define SEQ      1024
#define M_TOTAL  (BATCH*SEQ)   // 8192

// D(16x16 f32) = A(16x32 bf16) * B(32x16 bf16) + C
__device__ inline v8f wmma_bf16(v16bf a, v16bf b, v8f c) {
  return __builtin_amdgcn_wmma_f32_16x16x32_bf16(false, a, false, b, (short)0, c,
                                                 false, false);
}

__device__ inline v8f zero8() {
  v8f r;
#pragma unroll
  for (int i = 0; i < 8; ++i) r[i] = 0.0f;
  return r;
}

// A-matrix (16x32, 16-bit) lane layout: lanes 0-15 hold K slots {0..7,16..23},
// lanes 16-31 hold {8..15,24..31}. Caller passes p = row_base + (hi?8:0);
// the two 8-element chunks live at p and p+16. Both chunks are 16B-aligned.
__device__ inline v16bf load_a_frag(const __bf16* p) {
  v8bf c0 = *(const v8bf*)(p);
  v8bf c1 = *(const v8bf*)(p + 16);
  v16bf r;
#pragma unroll
  for (int i = 0; i < 8; ++i) { r[i] = c0[i]; r[i + 8] = c1[i]; }
  return r;
}

// ---------------- fp32 -> bf16 conversion (vectorized x4) ----------------
__global__ void __launch_bounds__(256)
cvt4_f32_bf16(const float* __restrict__ in, __bf16* __restrict__ out, int n4) {
  int i = blockIdx.x * blockDim.x + threadIdx.x;
  if (i < n4) {
    const float4 v = ((const float4*)in)[i];
    v4bf o;
    o[0] = (__bf16)v.x; o[1] = (__bf16)v.y; o[2] = (__bf16)v.z; o[3] = (__bf16)v.w;
    ((v4bf*)out)[i] = o;
  }
}

// ---------------- bf16 WMMA GEMM: C[M,N] = A[M,K] * B[K,N] + bias ----------------
// Workgroup: 128 threads = 4 waves; tile 128(M) x 64(N); K stepped by 32.
// Wave owns 32 rows: 2 A-frags x 4 shared B-frags = 8 WMMA per K-step.
// mode 0: store bf16 into head layout [B, H, S, D]   (for Q/K projections)
// mode 1: store fp32 into row-major [M, N]           (final output)
__global__ void __launch_bounds__(128)
gemm_bf16_wmma(const __bf16* __restrict__ A, const __bf16* __restrict__ B,
               const float* __restrict__ bias, void* __restrict__ Cout, int mode) {
  const int Kdim = N_HIDDEN, Ndim = N_HIDDEN;
  __shared__ __align__(32) __bf16 Ash[128][32];  // row-major, K contiguous (8 KB)
  __shared__ __align__(32) __bf16 Bt[64][32];    // Bt[n][k] = B[k][n]      (4 KB)

  const int m0 = blockIdx.x * 128;
  const int n0 = blockIdx.y * 64;
  const int t  = threadIdx.x;
  const int wid = t >> 5, lane = t & 31, lr = lane & 15;
  const bool hi = lane >= 16;

  v8f acc[2][4];
#pragma unroll
  for (int mf = 0; mf < 2; ++mf)
#pragma unroll
    for (int nb = 0; nb < 4; ++nb) acc[mf][nb] = zero8();

  const __bf16* arow = A + (size_t)(m0 + t) * Kdim;  // each thread stages one row

  for (int k0 = 0; k0 < Kdim; k0 += 32) {
    __syncthreads();
    {  // stage A tile: thread t copies its row's 32-element K-slab (64B)
      const v8bf* src = (const v8bf*)(arow + k0);
      v8bf c0 = src[0], c1 = src[1], c2 = src[2], c3 = src[3];
      *(v8bf*)&Ash[t][0]  = c0;  *(v8bf*)&Ash[t][8]  = c1;
      *(v8bf*)&Ash[t][16] = c2;  *(v8bf*)&Ash[t][24] = c3;
    }
    {  // stage B tile transposed so B-frags are contraction-contiguous per lane
      int kk = t >> 2, ng = t & 3;
      const __bf16* src = B + (size_t)(k0 + kk) * Ndim + n0 + ng * 16;
#pragma unroll
      for (int j = 0; j < 16; ++j) Bt[ng * 16 + j][kk] = src[j];
    }
    if (k0 + 32 < Kdim) {  // hint next K-slab into cache (global_prefetch_b8)
      __builtin_prefetch(arow + k0 + 32, 0, 1);
      __builtin_prefetch(B + (size_t)(k0 + 32 + (t >> 2)) * Ndim + n0 + (t & 3) * 16, 0, 1);
    }
    __syncthreads();

    v16bf a0 = load_a_frag(&Ash[wid * 32 + lr][hi ? 8 : 0]);
    v16bf a1 = load_a_frag(&Ash[wid * 32 + 16 + lr][hi ? 8 : 0]);
#pragma unroll
    for (int nb = 0; nb < 4; ++nb) {
      // B(32x16) lane layout: lanes 0-15 hold k={0..7,16..23}, lanes 16-31 hold {8..15,24..31}
      v16bf bfr = load_a_frag(&Bt[nb * 16 + lr][hi ? 8 : 0]);
      acc[0][nb] = wmma_bf16(a0, bfr, acc[0][nb]);
      acc[1][nb] = wmma_bf16(a1, bfr, acc[1][nb]);
    }
  }

  // epilogue: bias + store. C/D layout: lane -> col, vgpr(+half*8) -> row.
#pragma unroll
  for (int mf = 0; mf < 2; ++mf) {
#pragma unroll
    for (int nb = 0; nb < 4; ++nb) {
      const int col = n0 + nb * 16 + lr;
      const float bv = bias[col];
      const int rbase = m0 + wid * 32 + mf * 16 + (hi ? 8 : 0);
      if (mode == 0) {
        __bf16* out = (__bf16*)Cout;
        const int h = col >> 6, d = col & 63;
#pragma unroll
        for (int i = 0; i < 8; ++i) {
          int row = rbase + i;
          int bb = row >> 10, s = row & 1023;  // row = b*S + s
          out[(((size_t)(bb * N_HEAD + h)) * SEQ + s) * DIM_HEAD + d] =
              (__bf16)(acc[mf][nb][i] + bv);
        }
      } else {
        float* out = (float*)Cout;
#pragma unroll
        for (int i = 0; i < 8; ++i)
          out[(size_t)(rbase + i) * Ndim + col] = acc[mf][nb][i] + bv;
      }
    }
  }
}

// ---------------- flash-style attention (V = K per reference) ----------------
// grid: B*H*(S/64) blocks of 128 threads; wave w owns Q rows [qt*64+w*16, +16).
// K/V tile staged into LDS by the Tensor Data Mover (TDM).
__global__ void __launch_bounds__(128)
attn_wmma(const __bf16* __restrict__ Qh, const __bf16* __restrict__ Kh,
          const int* __restrict__ pmask, __bf16* __restrict__ att) {
  __shared__ __align__(32) __bf16 Ksh[32][64];     // [kv][d]  for QK^T B-frags
  __shared__ __align__(32) __bf16 Vt[64][32];      // [d][kv]  for P*V  B-frags
  __shared__ __align__(32) __bf16 Psh[4][16][32];  // per-wave P tile (A layout src)
  __shared__ float mrow[4][16];                    // per-wave running row max
  __shared__ float lrow[4][16];                    // per-wave running row sum

  const int bh = blockIdx.x >> 4;  // b*N_HEAD + h
  const int qt = blockIdx.x & 15;  // 64-row Q block index
  const int b  = bh >> 4;
  const int h  = bh & 15;
  const int t = threadIdx.x, wid = t >> 5, lane = t & 31, lr = lane & 15;
  const bool hi = lane >= 16;

  // Q fragments for this wave's 16 rows (held in registers for whole pass)
  const __bf16* qp =
      Qh + ((size_t)bh * SEQ + qt * 64 + wid * 16 + lr) * DIM_HEAD + (hi ? 8 : 0);
  v16bf qa0 = load_a_frag(qp);        // d = 0..31
  v16bf qa1 = load_a_frag(qp + 32);   // d = 32..63

  if (lane < 16) { mrow[wid][lane] = -3.0e38f; lrow[wid][lane] = 0.0f; }

  v8f acc[4];
#pragma unroll
  for (int nb = 0; nb < 4; ++nb) acc[nb] = zero8();

  const int* mg = pmask + b * SEQ;

  for (int kv0 = 0; kv0 < SEQ; kv0 += 32) {
    __syncthreads();  // everyone done reading previous tile
    if (t < 32) {
      // TDM: DMA the 32(kv) x 64(d) bf16 tile into Ksh.
      // D# per CDNA5 ISA: 2D tensor, data_size=2B,
      // tensor_dim0=64 (contiguous d), dim0_stride=64, tensor_dim1=1024 (kv),
      // tile = 64 x 32, global_addr = tile start, type=2 ("image").
      uint64_t ga = (uint64_t)(uintptr_t)(Kh + ((size_t)bh * SEQ + kv0) * DIM_HEAD);
      uint32_t la = (uint32_t)(uintptr_t)(&Ksh[0][0]);
      u32x4 g0;
      g0[0] = 1u;                                                  // count=1
      g0[1] = la;                                                  // lds_addr
      g0[2] = (uint32_t)ga;                                        // global_addr lo
      g0[3] = (uint32_t)((ga >> 32) & 0x01FFFFFFu) | 0x80000000u;  // addr hi | type=2
      i32x8 g1;
      g1[0] = 0x00010000;        // workgroup_mask=0, data_size=1 (2 bytes)
      g1[1] = (int)(64u << 16);  // tensor_dim0 = 64  (bits 63:48)
      g1[2] = (int)(1024u << 16);// tensor_dim1 = 1024 (bits 95:80)
      g1[3] = (int)(64u << 16);  // tile_dim0 = 64    (bits 127:112)
      g1[4] = 32;                // tile_dim1 = 32    (bits 143:128)
      g1[5] = 64;                // tensor_dim0_stride = 64 (bits 191:160)
      g1[6] = 0;
      g1[7] = 0;
      i32x4 z4; z4[0] = 0; z4[1] = 0; z4[2] = 0; z4[3] = 0;
      i32x8 z8;
#pragma unroll
      for (int j = 0; j < 8; ++j) z8[j] = 0;
      // 6-arg form (clang-23 / therock-10.0 headers): extra int32x8 group + cpol
      __builtin_amdgcn_tensor_load_to_lds(g0, g1, z4, z4, z8, 0);
      __builtin_amdgcn_s_wait_tensorcnt(0);
    }
    __syncthreads();  // Ksh visible to all waves
    {  // build d-major copy Vt from Ksh (K doubles as V)
      int r = t >> 2, dq = (t & 3) * 16;
      v8bf c0 = *(const v8bf*)&Ksh[r][dq];
      v8bf c1 = *(const v8bf*)&Ksh[r][dq + 8];
#pragma unroll
      for (int j = 0; j < 8; ++j) { Vt[dq + j][r] = c0[j]; Vt[dq + 8 + j][r] = c1[j]; }
    }
    __syncthreads();

    // ---- logits = Q * K^T : 16x32 in two 16x16 frags, contraction d=64 ----
    v8f f0 = zero8(), f1 = zero8();
    {
      v16bf b00 = load_a_frag(&Ksh[lr][hi ? 8 : 0]);
      v16bf b01 = load_a_frag(&Ksh[lr][32 + (hi ? 8 : 0)]);
      v16bf b10 = load_a_frag(&Ksh[16 + lr][hi ? 8 : 0]);
      v16bf b11 = load_a_frag(&Ksh[16 + lr][32 + (hi ? 8 : 0)]);
      f0 = wmma_bf16(qa0, b00, f0);  f0 = wmma_bf16(qa1, b01, f0);
      f1 = wmma_bf16(qa0, b10, f1);  f1 = wmma_bf16(qa1, b11, f1);
    }

    // key padding mask (column-dependent only)
    const float mv0 = -1.0e9f * (float)mg[kv0 + lr];
    const float mv1 = -1.0e9f * (float)mg[kv0 + 16 + lr];

    // ---- online softmax: rows live across the 16-lane halves ----
    float sc[8];
#pragma unroll
    for (int i = 0; i < 8; ++i) {
      float a0 = f0[i] * 0.125f + mv0;   // 1/sqrt(64)
      float a1 = f1[i] * 0.125f + mv1;
      float rm = fmaxf(a0, a1);
#pragma unroll
      for (int off = 1; off < 16; off <<= 1) rm = fmaxf(rm, __shfl_xor(rm, off, 32));
      const int row = i + (hi ? 8 : 0);
      float om = mrow[wid][row];
      float nm = fmaxf(om, rm);
      sc[i] = __expf(om - nm);
      float p0 = __expf(a0 - nm);
      float p1 = __expf(a1 - nm);
      float rs = p0 + p1;
#pragma unroll
      for (int off = 1; off < 16; off <<= 1) rs += __shfl_xor(rs, off, 32);
      if (lr == 0) {  // lane0 -> rows 0..7, lane16 -> rows 8..15
        lrow[wid][row] = lrow[wid][row] * sc[i] + rs;
        mrow[wid][row] = nm;
      }
      // park P in LDS to re-shape D-layout -> A-layout for the P*V WMMA
      Psh[wid][row][lr]      = (__bf16)p0;
      Psh[wid][row][16 + lr] = (__bf16)p1;
    }

    // rescale running accumulator by exp(m_old - m_new) per row
#pragma unroll
    for (int nb = 0; nb < 4; ++nb)
#pragma unroll
      for (int i = 0; i < 8; ++i) acc[nb][i] *= sc[i];

    // ---- att += P(16x32) * V(32x64), V == K tile ----
    v16bf pa = load_a_frag(&Psh[wid][lr][hi ? 8 : 0]);
#pragma unroll
    for (int nb = 0; nb < 4; ++nb) {
      v16bf vb = load_a_frag(&Vt[nb * 16 + lr][hi ? 8 : 0]);
      acc[nb] = wmma_bf16(pa, vb, acc[nb]);
    }
  }

  // normalize by row sums and store att as bf16 in [B, S, H*D]
  float inv[8];
#pragma unroll
  for (int i = 0; i < 8; ++i) inv[i] = 1.0f / lrow[wid][i + (hi ? 8 : 0)];
#pragma unroll
  for (int nb = 0; nb < 4; ++nb) {
    const int col = h * DIM_HEAD + nb * 16 + lr;
#pragma unroll
    for (int i = 0; i < 8; ++i) {
      const int s = qt * 64 + wid * 16 + i + (hi ? 8 : 0);
      att[((size_t)b * SEQ + s) * N_HIDDEN + col] = (__bf16)(acc[nb][i] * inv[i]);
    }
  }
}

// -------------------------------- launcher --------------------------------
extern "C" void kernel_launch(void* const* d_in, const int* in_sizes, int n_in,
                              void* d_out, int out_size, void* d_ws, size_t ws_size,
                              hipStream_t stream) {
  (void)in_sizes; (void)n_in; (void)out_size; (void)ws_size;
  const float* x  = (const float*)d_in[0];
  const int*   pm = (const int*)d_in[1];
  const float* Wq = (const float*)d_in[2];
  const float* bq = (const float*)d_in[3];
  const float* Wk = (const float*)d_in[4];
  const float* bk = (const float*)d_in[5];
  const float* Wo = (const float*)d_in[6];
  const float* bo = (const float*)d_in[7];
  float* out = (float*)d_out;

  // workspace layout (bf16 elements), ~73 MB total
  const size_t XE = (size_t)M_TOTAL * N_HIDDEN;   // 8.39M
  const size_t WE = (size_t)N_HIDDEN * N_HIDDEN;  // 1.05M
  __bf16* ws  = (__bf16*)d_ws;
  __bf16* xb  = ws;            // x bf16
  __bf16* wqb = xb + XE;
  __bf16* wkb = wqb + WE;
  __bf16* wob = wkb + WE;
  __bf16* qh  = wob + WE;      // Q [B,H,S,D]
  __bf16* kh  = qh + XE;       // K [B,H,S,D]  (also serves as V)
  __bf16* ab  = kh + XE;       // att [B,S,N_HIDDEN]

  cvt4_f32_bf16<<<(int)(XE / 4 / 256), 256, 0, stream>>>(x,  xb,  (int)(XE / 4));
  cvt4_f32_bf16<<<(int)(WE / 4 / 256), 256, 0, stream>>>(Wq, wqb, (int)(WE / 4));
  cvt4_f32_bf16<<<(int)(WE / 4 / 256), 256, 0, stream>>>(Wk, wkb, (int)(WE / 4));
  cvt4_f32_bf16<<<(int)(WE / 4 / 256), 256, 0, stream>>>(Wo, wob, (int)(WE / 4));

  dim3 gg(M_TOTAL / 128, N_HIDDEN / 64);  // 64 x 16
  gemm_bf16_wmma<<<gg, 128, 0, stream>>>(xb, wqb, bq, qh, 0);
  gemm_bf16_wmma<<<gg, 128, 0, stream>>>(xb, wkb, bk, kh, 0);

  attn_wmma<<<BATCH * N_HEAD * (SEQ / 64), 128, 0, stream>>>(qh, kh, pm, ab);

  gemm_bf16_wmma<<<gg, 128, 0, stream>>>(ab, wob, bo, (void*)out, 1);
}
